// MultiHeadSelfAttention_9801115370085
// MI455X (gfx1250) — matrix operand order repaired and checked by reading, not compiled, not run on hardware
//
#include <hip/hip_runtime.h>

#define EMBED 1024
#define NHEADS 16
#define HDIM 64
#define SEQ 2048
#define BATCH 2

typedef __attribute__((ext_vector_type(16))) __bf16 v16bf;
typedef __attribute__((ext_vector_type(8))) float v8f;

union FragB16 {
  v16bf v;
  uint4 q[2];
};

__device__ __forceinline__ unsigned short f32_to_bf16(float f) {
  union { float f; unsigned int u; } v;
  v.f = f;
  unsigned int r = v.u + 0x7FFFu + ((v.u >> 16) & 1u);
  return (unsigned short)(r >> 16);
}

__device__ __forceinline__ v8f vzero8() {
  v8f z;
#pragma unroll
  for (int i = 0; i < 8; ++i) z[i] = 0.0f;
  return z;
}

__device__ __forceinline__ v8f wmma_bf16(const FragB16& a, const FragB16& b, v8f c) {
  return __builtin_amdgcn_wmma_f32_16x16x32_bf16(false, a.v, false, b.v, (short)0, c,
                                                 false, false);
}

// CDNA5 async copy: global -> LDS, 16 bytes per lane, tracked on ASYNCcnt.
// GVS addressing: mem = SADDR(64b sgpr) + VADDR(32b byte offset); LDS addr in VGPR.
__device__ __forceinline__ void async_load_b128(unsigned lds_addr, unsigned goff,
                                                const void* base) {
  asm volatile("global_load_async_to_lds_b128 %0, %1, %2"
               :
               : "v"(lds_addr), "v"(goff), "s"(base)
               : "memory");
}

__device__ __forceinline__ void wait_async0() {
  asm volatile("s_wait_asynccnt 0x0" ::: "memory");
}

__device__ __forceinline__ unsigned lds_addr_of(const void* p) {
  return (unsigned)(size_t)p;  // low 32 bits of flat shared address = LDS offset
}

// ---------------------------------------------------------------------------
// One-time elementwise fp32 -> bf16 pack (4 elems/thread)
// ---------------------------------------------------------------------------
__global__ __launch_bounds__(256) void convert_bf16_kernel(
    const float* __restrict__ src, unsigned short* __restrict__ dst) {
  size_t i = ((size_t)blockIdx.x * 256 + threadIdx.x) * 4;
  float4 f = *(const float4*)&src[i];
  uint2 p;
  p.x = (unsigned)f32_to_bf16(f.x) | ((unsigned)f32_to_bf16(f.y) << 16);
  p.y = (unsigned)f32_to_bf16(f.z) | ((unsigned)f32_to_bf16(f.w) << 16);
  *(uint2*)&dst[i] = p;
}

// ---------------------------------------------------------------------------
// One-time fp32 [K][N] -> bf16 transposed [N][K] (32x32 LDS tile)
// ---------------------------------------------------------------------------
__global__ __launch_bounds__(256) void transpose_convert_kernel(
    const float* __restrict__ W, unsigned short* __restrict__ Wt, int K, int N) {
  __shared__ unsigned short t[32][33];
  const int tx = threadIdx.x & 31;
  const int ty = threadIdx.x >> 5;
  const int n0 = blockIdx.x * 32;
  const int k0 = blockIdx.y * 32;
#pragma unroll
  for (int i = 0; i < 4; ++i) {
    int r = ty + i * 8;
    t[r][tx] = f32_to_bf16(W[(size_t)(k0 + r) * N + n0 + tx]);
  }
  __syncthreads();
#pragma unroll
  for (int i = 0; i < 4; ++i) {
    int r = ty + i * 8;
    Wt[(size_t)(n0 + r) * K + k0 + tx] = t[tx][r];
  }
}

// ---------------------------------------------------------------------------
// QKV GEMM: bf16 Xb[4096,1024] x bf16 Wt[3072,1024](pre-transposed) + bias
// -> Q/K bf16 [B,H,N,D] (Q pre-scaled 1/sqrt(D)), V bf16 transposed [B,H,D,N].
// 128x128x32 tiles, async-to-LDS staging, 8 waves, 2x4 subtiles/wave.
// ---------------------------------------------------------------------------
__global__ __launch_bounds__(256) void qkv_gemm_kernel(
    const unsigned short* __restrict__ Xb, const unsigned short* __restrict__ Wt,
    const float* __restrict__ bias, unsigned short* __restrict__ Qo,
    unsigned short* __restrict__ Ko, unsigned short* __restrict__ Vt) {
  __shared__ unsigned short sA[128 * 32];  // [m][k]
  __shared__ unsigned short sB[128 * 32];  // [n][k]

  const int tid = threadIdx.x;
  const int wave = tid >> 5;
  const int lane = tid & 31;
  const int half = lane >> 4;
  const int lr = lane & 15;
  const int sr = (wave >> 1) << 1;
  const int sc = (wave & 1) << 2;
  const int mBase = blockIdx.y * 128;
  const int nBase = blockIdx.x * 128;

  // async staging: 512 x 16B chunks per tile, 2 per thread per tile
  unsigned aLds[2], bLds[2], aOff[2], bOff[2];
#pragma unroll
  for (int i = 0; i < 2; ++i) {
    int idx = tid + i * 256;
    int row = idx >> 2;
    int kcc = (idx & 3) << 3;
    aLds[i] = lds_addr_of(&sA[row * 32 + kcc]);
    bLds[i] = lds_addr_of(&sB[row * 32 + kcc]);
    aOff[i] = (unsigned)(((mBase + row) * EMBED + kcc) * 2);
    bOff[i] = (unsigned)(((nBase + row) * EMBED + kcc) * 2);
  }

  v8f acc[2][4];
#pragma unroll
  for (int r = 0; r < 2; ++r)
#pragma unroll
    for (int c = 0; c < 4; ++c) acc[r][c] = vzero8();

  for (int kb = 0; kb < EMBED; kb += 32) {
    __syncthreads();
#pragma unroll
    for (int i = 0; i < 2; ++i) async_load_b128(aLds[i], aOff[i] + kb * 2, Xb);
#pragma unroll
    for (int i = 0; i < 2; ++i) async_load_b128(bLds[i], bOff[i] + kb * 2, Wt);
    wait_async0();
    __syncthreads();

    FragB16 fa[2], fb[4];
#pragma unroll
    for (int r = 0; r < 2; ++r) {
      int m0 = (sr + r) * 16;
      int k0 = half * 8;
      fa[r].q[0] = *(const uint4*)&sA[(m0 + lr) * 32 + k0];
      fa[r].q[1] = *(const uint4*)&sA[(m0 + lr) * 32 + k0 + 16];
    }
#pragma unroll
    for (int c = 0; c < 4; ++c) {
      int n0 = (sc + c) * 16;
      fb[c].q[0] = *(const uint4*)&sB[(n0 + lr) * 32 + half * 8];
      fb[c].q[1] = *(const uint4*)&sB[(n0 + lr) * 32 + half * 8 + 16];
    }
#pragma unroll
    for (int r = 0; r < 2; ++r)
#pragma unroll
      for (int c = 0; c < 4; ++c) acc[r][c] = wmma_bf16(fa[r], fb[c], acc[r][c]);
  }

  // epilogue: Q/K -> [B,H,N,D]; V -> transposed [B,H,D,N]
#pragma unroll
  for (int r = 0; r < 2; ++r) {
#pragma unroll
    for (int c = 0; c < 4; ++c) {
      int col = nBase + (sc + c) * 16 + lr;
      int which = col >> 10;
      int e = col & 1023;
      int h = e >> 6;
      int d = e & 63;
      float bv = bias[col];
      float scl = (which == 0) ? 0.125f : 1.0f;  // 1/sqrt(64) folded into Q
#pragma unroll
      for (int j = 0; j < 8; ++j) {
        int m = mBase + (sr + r) * 16 + j + 8 * half;
        int bb = m >> 11;
        int n = m & (SEQ - 1);
        unsigned short val = f32_to_bf16((acc[r][c][j] + bv) * scl);
        if (which == 2) {
          Vt[(size_t)((bb * NHEADS + h) * HDIM + d) * SEQ + n] = val;
        } else {
          unsigned short* dst = (which == 0) ? Qo : Ko;
          dst[(size_t)(((bb * NHEADS + h) * SEQ) + n) * HDIM + d] = val;
        }
      }
    }
  }
}

// ---------------------------------------------------------------------------
// Flash attention: per (b,h), one 16-row query tile per wave, KV chunks of 32.
// K and (pre-transposed) V chunks staged with async-to-LDS; S=QK^T and O+=PV
// via bf16 WMMA; online softmax in the WMMA C-layout.
// ---------------------------------------------------------------------------
__global__ __launch_bounds__(256) void attn_kernel(
    const unsigned short* __restrict__ Qg, const unsigned short* __restrict__ Kg,
    const unsigned short* __restrict__ Vtg, unsigned short* __restrict__ Og) {
  __shared__ unsigned short sK[32 * 64];     // [key][d]
  __shared__ unsigned short sVt[64 * 32];    // [d][key]
  __shared__ unsigned short sP[8][16 * 32];  // per-wave P tile

  const int tid = threadIdx.x;
  const int wave = tid >> 5;
  const int lane = tid & 31;
  const int half = lane >> 4;
  const int lr = lane & 15;

  const int bh = blockIdx.x;  // 0..31
  const int b = bh >> 4;
  const int h = bh & 15;
  const int q0 = blockIdx.y * 128 + wave * 16;

  const unsigned short* Qp = Qg + (size_t)bh * SEQ * HDIM;
  const unsigned short* Kp = Kg + (size_t)bh * SEQ * HDIM;
  const unsigned short* Vtp = Vtg + (size_t)bh * HDIM * SEQ;

  // Q fragments: two k-steps across D=64
  FragB16 qf[2];
#pragma unroll
  for (int s = 0; s < 2; ++s) {
    int kd = s * 32 + half * 8;
    qf[s].q[0] = *(const uint4*)&Qp[(size_t)(q0 + lr) * HDIM + kd];
    qf[s].q[1] = *(const uint4*)&Qp[(size_t)(q0 + lr) * HDIM + kd + 16];
  }

  // async staging descriptors (one 16B chunk per thread per buffer)
  const unsigned kLds = lds_addr_of(&sK[tid * 8]);
  const unsigned kOffBase = (unsigned)(tid * 16);  // bytes
  const int vd = tid >> 2;
  const int vn = (tid & 3) << 3;
  const unsigned vLds = lds_addr_of(&sVt[vd * 32 + vn]);
  const unsigned vOffBase = (unsigned)((vd * SEQ + vn) * 2);

  float mrow[8], lrow[8];
  v8f oacc[4];
#pragma unroll
  for (int j = 0; j < 8; ++j) {
    mrow[j] = -3.0e38f;
    lrow[j] = 0.0f;
  }
#pragma unroll
  for (int dt = 0; dt < 4; ++dt) oacc[dt] = vzero8();

  for (int kc = 0; kc < SEQ; kc += 32) {
    __syncthreads();
    async_load_b128(kLds, kOffBase + (unsigned)kc * 128u, Kp);
    async_load_b128(vLds, vOffBase + (unsigned)kc * 2u, Vtp);
    wait_async0();
    __syncthreads();

    // S = Q K^T : two 16x16 key tiles, two k-steps each over D
    v8f st[2];
#pragma unroll
    for (int t = 0; t < 2; ++t) {
      st[t] = vzero8();
#pragma unroll
      for (int s = 0; s < 2; ++s) {
        FragB16 fb;
        int kb = s * 32 + half * 8;
        fb.q[0] = *(const uint4*)&sK[(t * 16 + lr) * 64 + kb];
        fb.q[1] = *(const uint4*)&sK[(t * 16 + lr) * 64 + kb + 16];
        st[t] = wmma_bf16(qf[s], fb, st[t]);
      }
    }

    // online softmax: row = j + 8*half, lane owns col lr (+16 for tile 1)
#pragma unroll
    for (int j = 0; j < 8; ++j) {
      float cm = fmaxf(st[0][j], st[1][j]);
#pragma unroll
      for (int off = 1; off < 16; off <<= 1) cm = fmaxf(cm, __shfl_xor(cm, off, 32));
      float nm = fmaxf(mrow[j], cm);
      float corr = __expf(mrow[j] - nm);
      mrow[j] = nm;
      float p0 = __expf(st[0][j] - nm);
      float p1 = __expf(st[1][j] - nm);
      float rs = p0 + p1;
#pragma unroll
      for (int off = 1; off < 16; off <<= 1) rs += __shfl_xor(rs, off, 32);
      lrow[j] = lrow[j] * corr + rs;
      int row = j + 8 * half;
      sP[wave][row * 32 + lr] = f32_to_bf16(p0);
      sP[wave][row * 32 + 16 + lr] = f32_to_bf16(p1);
#pragma unroll
      for (int dt = 0; dt < 4; ++dt) oacc[dt][j] *= corr;
    }

    // P as A-fragment (16x32)
    FragB16 pf;
    {
      int k0 = half * 8;
      pf.q[0] = *(const uint4*)&sP[wave][lr * 32 + k0];
      pf.q[1] = *(const uint4*)&sP[wave][lr * 32 + k0 + 16];
    }
    // O += P V
#pragma unroll
    for (int dt = 0; dt < 4; ++dt) {
      FragB16 fv;
      fv.q[0] = *(const uint4*)&sVt[(dt * 16 + lr) * 32 + half * 8];
      fv.q[1] = *(const uint4*)&sVt[(dt * 16 + lr) * 32 + half * 8 + 16];
      oacc[dt] = wmma_bf16(pf, fv, oacc[dt]);
    }
  }

  // epilogue: normalize, store bf16 attention output [B][N][E]
#pragma unroll
  for (int j = 0; j < 8; ++j) {
    float inv = 1.0f / lrow[j];
    int n = q0 + j + 8 * half;
#pragma unroll
    for (int dt = 0; dt < 4; ++dt) {
      int d = dt * 16 + lr;
      Og[(size_t)(b * SEQ + n) * EMBED + h * HDIM + d] = f32_to_bf16(oacc[dt][j] * inv);
    }
  }
}

// ---------------------------------------------------------------------------
// Output projection: bf16 [4096,1024] x bf16 Wt[1024,1024](pre-transposed)
// + bias -> f32 out. Async-to-LDS staging for both tiles.
// ---------------------------------------------------------------------------
__global__ __launch_bounds__(256) void proj_gemm_kernel(
    const unsigned short* __restrict__ A, const unsigned short* __restrict__ Wt,
    const float* __restrict__ bias, float* __restrict__ out) {
  const int N = EMBED;
  __shared__ unsigned short sA[128 * 32];
  __shared__ unsigned short sB[128 * 32];

  const int tid = threadIdx.x;
  const int wave = tid >> 5;
  const int lane = tid & 31;
  const int half = lane >> 4;
  const int lr = lane & 15;
  const int sr = (wave >> 1) << 1;
  const int sc = (wave & 1) << 2;
  const int mBase = blockIdx.y * 128;
  const int nBase = blockIdx.x * 128;

  unsigned aLds[2], bLds[2], aOff[2], bOff[2];
#pragma unroll
  for (int i = 0; i < 2; ++i) {
    int idx = tid + i * 256;
    int row = idx >> 2;
    int kcc = (idx & 3) << 3;
    aLds[i] = lds_addr_of(&sA[row * 32 + kcc]);
    bLds[i] = lds_addr_of(&sB[row * 32 + kcc]);
    aOff[i] = (unsigned)(((mBase + row) * EMBED + kcc) * 2);
    bOff[i] = (unsigned)(((nBase + row) * EMBED + kcc) * 2);
  }

  v8f acc[2][4];
#pragma unroll
  for (int r = 0; r < 2; ++r)
#pragma unroll
    for (int c = 0; c < 4; ++c) acc[r][c] = vzero8();

  for (int kb = 0; kb < EMBED; kb += 32) {
    __syncthreads();
#pragma unroll
    for (int i = 0; i < 2; ++i) async_load_b128(aLds[i], aOff[i] + kb * 2, A);
#pragma unroll
    for (int i = 0; i < 2; ++i) async_load_b128(bLds[i], bOff[i] + kb * 2, Wt);
    wait_async0();
    __syncthreads();

    FragB16 fa[2], fb[4];
#pragma unroll
    for (int r = 0; r < 2; ++r) {
      int m0 = (sr + r) * 16;
      int k0 = half * 8;
      fa[r].q[0] = *(const uint4*)&sA[(m0 + lr) * 32 + k0];
      fa[r].q[1] = *(const uint4*)&sA[(m0 + lr) * 32 + k0 + 16];
    }
#pragma unroll
    for (int c = 0; c < 4; ++c) {
      int n0 = (sc + c) * 16;
      fb[c].q[0] = *(const uint4*)&sB[(n0 + lr) * 32 + half * 8];
      fb[c].q[1] = *(const uint4*)&sB[(n0 + lr) * 32 + half * 8 + 16];
    }
#pragma unroll
    for (int r = 0; r < 2; ++r)
#pragma unroll
      for (int c = 0; c < 4; ++c) acc[r][c] = wmma_bf16(fa[r], fb[c], acc[r][c]);
  }

#pragma unroll
  for (int r = 0; r < 2; ++r) {
#pragma unroll
    for (int c = 0; c < 4; ++c) {
      int col = nBase + (sc + c) * 16 + lr;
      float bv = bias[col];
#pragma unroll
      for (int j = 0; j < 8; ++j) {
        int m = mBase + (sr + r) * 16 + j + 8 * half;
        out[(size_t)m * N + col] = acc[r][c][j] + bv;
      }
    }
  }
}

extern "C" void kernel_launch(void* const* d_in, const int* in_sizes, int n_in,
                              void* d_out, int out_size, void* d_ws, size_t ws_size,
                              hipStream_t stream) {
  const float* x = (const float*)d_in[0];
  const float* w_qkv = (const float*)d_in[1];
  const float* b_qkv = (const float*)d_in[2];
  const float* w_proj = (const float*)d_in[3];
  const float* b_proj = (const float*)d_in[4];
  float* out = (float*)d_out;

  const size_t nX = (size_t)BATCH * SEQ * EMBED;         // 4,194,304
  const size_t nWqkv = (size_t)EMBED * 3 * EMBED;        // 3,145,728
  const size_t nWproj = (size_t)EMBED * EMBED;           // 1,048,576
  const size_t per = (size_t)BATCH * NHEADS * SEQ * HDIM;  // 4,194,304

  unsigned short* xb = (unsigned short*)d_ws;
  unsigned short* wqkvt = xb + nX;
  unsigned short* wprojt = wqkvt + nWqkv;
  unsigned short* qs = wprojt + nWproj;
  unsigned short* ks = qs + per;
  unsigned short* vt = ks + per;
  unsigned short* os = vt + per;

  dim3 blk(256, 1, 1);
  convert_bf16_kernel<<<(unsigned)(nX / 1024), blk, 0, stream>>>(x, xb);
  transpose_convert_kernel<<<dim3(3 * EMBED / 32, EMBED / 32), blk, 0, stream>>>(
      w_qkv, wqkvt, EMBED, 3 * EMBED);
  transpose_convert_kernel<<<dim3(EMBED / 32, EMBED / 32), blk, 0, stream>>>(
      w_proj, wprojt, EMBED, EMBED);
  qkv_gemm_kernel<<<dim3(3 * EMBED / 128, BATCH * SEQ / 128), blk, 0, stream>>>(
      xb, wqkvt, b_qkv, qs, ks, vt);
  attn_kernel<<<dim3(BATCH * NHEADS, SEQ / 128), blk, 0, stream>>>(qs, ks, vt, os);
  proj_gemm_kernel<<<dim3(EMBED / 128, BATCH * SEQ / 128), blk, 0, stream>>>(
      os, wprojt, b_proj, out);
}
